// S6_75539884802689
// MI455X (gfx1250) — matrix operand order repaired and checked by reading, not compiled, not run on hardware
//
#include <hip/hip_runtime.h>
#include <cstdint>
#include <cstddef>

typedef __attribute__((ext_vector_type(16))) _Float16 v16h;
typedef __attribute__((ext_vector_type(8)))  float    v8f;

namespace {
constexpr int kB  = 4;
constexpr int kL  = 2048;
constexpr int kF  = 128;
constexpr int kN  = 32;
constexpr int kML = kB * kL;   // 8192 (b,l) rows
constexpr float kDelta0 = 1.0f;
}

// ---------------- Phase 0: weight pre-swizzle into WMMA B layout ----------------
// wsWh[(s*32+lane)*48 + {0..15 | 16..31 | 32..47}] holds the v16h B operands
// (tile n=0..15, tile n=16..31, sD_w column tile) for K-step s, lane `lane`.
// 16-bit B-matrix 32x16 layout: lane column = lane&15; element e of the v16h
// maps to K = s*32 + hi*8 + e (e<8), K = s*32 + 16 + hi*8 + (e-8) (e>=8).
__global__ __launch_bounds__(128) void s6_wconv(
    const float* __restrict__ sBw, const float* __restrict__ sDw,
    _Float16* __restrict__ wsWh) {
  const int t    = threadIdx.x;    // 0..127 = s*32 + lane
  const int s    = t >> 5;
  const int lane = t & 31;
  const int m    = lane & 15;
  const int hi   = lane >> 4;
  _Float16* dst = wsWh + (size_t)t * 48;
#pragma unroll
  for (int e = 0; e < 16; ++e) {
    const int k = s * 32 + (e < 8 ? hi * 8 + e : 16 + hi * 8 + (e - 8));
    dst[e]      = (_Float16)sBw[m * kF + k];          // B[k][n] = sB_w[n][k]
    dst[16 + e] = (_Float16)sBw[(m + 16) * kF + k];
    dst[32 + e] = (m == 0) ? (_Float16)sDw[k] : (_Float16)0.0f;
  }
}

// ---------------- Phase 1: WMMA projections ----------------
// One wave per 16-row tile: Bm[rows,0:32] = x . sB_w^T (two 16x16 tiles) and
// d1raw[rows] = x . sD_w (third tile, column 0). Weights come pre-swizzled
// from wsWh as contiguous v16h; A tiles as float4 pairs -> packed f16.
__global__ __launch_bounds__(32) void s6_proj(
    const float* __restrict__ x, const _Float16* __restrict__ wsWh,
    float* __restrict__ wsBm, float* __restrict__ wsD1) {
  const int lid  = threadIdx.x;
  const int m    = lid & 15;   // A-row / D-column index
  const int hi   = lid >> 4;
  const int row0 = blockIdx.x * 16;

  v8f acc0 = {}, acc1 = {}, acc2 = {};
#pragma unroll
  for (int s = 0; s < 4; ++s) {          // K = 128 in 4 steps of 32
    // A operand: 16-bit A 16x32 layout -> lane needs two contiguous 8-float
    // runs of its row: k in [s*32+hi*8, +8) and [s*32+16+hi*8, +8).
    const float* xrow = x + (size_t)(row0 + m) * kF + s * 32 + hi * 8;
    const float4 r0 = ((const float4*)xrow)[0];
    const float4 r1 = ((const float4*)xrow)[1];
    const float4 r2 = ((const float4*)(xrow + 16))[0];
    const float4 r3 = ((const float4*)(xrow + 16))[1];
    v16h a;
    a[0]=(_Float16)r0.x; a[1]=(_Float16)r0.y; a[2]=(_Float16)r0.z; a[3]=(_Float16)r0.w;
    a[4]=(_Float16)r1.x; a[5]=(_Float16)r1.y; a[6]=(_Float16)r1.z; a[7]=(_Float16)r1.w;
    a[8]=(_Float16)r2.x; a[9]=(_Float16)r2.y; a[10]=(_Float16)r2.z; a[11]=(_Float16)r2.w;
    a[12]=(_Float16)r3.x; a[13]=(_Float16)r3.y; a[14]=(_Float16)r3.z; a[15]=(_Float16)r3.w;

    const v16h* wp = (const v16h*)(wsWh + (size_t)(s * 32 + lid) * 48);
    const v16h t0 = wp[0];
    const v16h t1 = wp[1];
    const v16h t2 = wp[2];

    acc0 = __builtin_amdgcn_wmma_f32_16x16x32_f16(false, a, false, t0, (short)0, acc0, false, false);
    acc1 = __builtin_amdgcn_wmma_f32_16x16x32_f16(false, a, false, t1, (short)0, acc1, false, false);
    acc2 = __builtin_amdgcn_wmma_f32_16x16x32_f16(false, a, false, t2, (short)0, acc2, false, false);
  }
  // D layout: VGPR v -> row M = v + 8*hi, column = lane&15.
#pragma unroll
  for (int v = 0; v < 8; ++v) {
    const int r = row0 + v + 8 * hi;
    wsBm[(size_t)r * kN + m]      = acc0[v];
    wsBm[(size_t)r * kN + m + 16] = acc1[v];
    if (m == 0) wsD1[r] = acc2[v];
  }
}

// ---------------- Phase 2: elementwise Ad / Bd ----------------
__global__ __launch_bounds__(256) void s6_adbd(
    const float* __restrict__ wsBm, const float* __restrict__ wsD1,
    const float* __restrict__ Aev,   // d_in[1]: complex64 (F,N); row 0 suffices
    const float* __restrict__ sBb, const float* __restrict__ sDb,
    float* __restrict__ wsAB) {
  const int idx = blockIdx.x * blockDim.x + threadIdx.x;
  if (idx >= kML * kN) return;
  const int n = idx & (kN - 1);
  const int r = idx >> 5;

  const float z = kDelta0 + wsD1[r] + sDb[0];
  const float delta = (z > 20.0f) ? z : log1pf(expf(z));   // softplus

  const float evr = Aev[2 * n], evi = Aev[2 * n + 1];      // A[0][n]
  const float dAr = delta * evr, dAi = delta * evi;
  const float e   = expf(dAr);
  const float Adr = e * cosf(dAi);
  const float Adi = e * sinf(dAi);

  const float inv = 1.0f / (dAr * dAr + dAi * dAi);        // 1/dA = conj/|dA|^2
  const float fr  = 1.0f - dAr * inv;                      // (dA-1)/dA
  const float fi  = dAi * inv;
  const float u   = delta * (wsBm[idx] + sBb[n]);

  ((float4*)wsAB)[idx] = make_float4(Adr, Adi, fr * u, fi * u);
}

// ---------------- Phase 3: suffix scan -> g[b,l] ----------------
__global__ __launch_bounds__(128) void s6_scan(
    const float* __restrict__ x, const float* __restrict__ sCw,
    const float* __restrict__ sCb, const float* __restrict__ wsAB,
    float* __restrict__ wsG) {
  const int tid = threadIdx.x;
  const int b = tid >> 5;
  const int n = tid & 31;

  // Cl[b,n] = x[b,L-1,:] . sC_w[n,:] + sC_b[n]  (imag part is zero)
  const float* xl = x + ((size_t)b * kL + (kL - 1)) * kF;
  float cl = sCb[n];
  for (int f = 0; f < kF; ++f) cl = fmaf(xl[f], sCw[n * kF + f], cl);

  const float4* ab4 = (const float4*)wsAB;
  float Pr = 1.0f, Pi = 0.0f;
  for (int l = kL - 1; l >= 0; --l) {
    const size_t base = ((size_t)b * kL + l) * kN + n;
    if (l >= 16) __builtin_prefetch(&ab4[base - 16 * kN], 0, 1);
    const float4 ab = ab4[base];
    // w = P * Bd ; contribution = Cl * w (Cl real)
    float gr = cl * (Pr * ab.z - Pi * ab.w);
    float gi = cl * (Pr * ab.w + Pi * ab.z);
#pragma unroll
    for (int off = 16; off > 0; off >>= 1) {
      gr += __shfl_xor(gr, off, 32);
      gi += __shfl_xor(gi, off, 32);
    }
    if (n == 0) {
      wsG[2 * ((size_t)b * kL + l)]     = gr;
      wsG[2 * ((size_t)b * kL + l) + 1] = gi;
    }
    const float npr = Pr * ab.x - Pi * ab.y;   // P *= Ad
    const float npi = Pr * ab.y + Pi * ab.x;
    Pr = npr; Pi = npi;
  }
}

// ---------------- Phase 4: y[b,f] = sum_l g[b,l] * x[b,l,f] ----------------
__global__ __launch_bounds__(128) void s6_out(
    const float* __restrict__ x, const float* __restrict__ wsG,
    float* __restrict__ out) {
  const int b = blockIdx.x;
  const int f = threadIdx.x;
  const float* xb = x + (size_t)b * kL * kF;
  float ar = 0.0f, ai = 0.0f;
  for (int l = 0; l < kL; ++l) {
    const float xv = xb[(size_t)l * kF + f];
    ar = fmaf(wsG[2 * ((size_t)b * kL + l)],     xv, ar);
    ai = fmaf(wsG[2 * ((size_t)b * kL + l) + 1], xv, ai);
  }
  out[2 * (b * kF + f)]     = ar;   // complex64 interleaved
  out[2 * (b * kF + f) + 1] = ai;
}

extern "C" void kernel_launch(void* const* d_in, const int* in_sizes, int n_in,
                              void* d_out, int out_size, void* d_ws, size_t ws_size,
                              hipStream_t stream) {
  (void)in_sizes; (void)n_in; (void)out_size; (void)ws_size;
  const float* x   = (const float*)d_in[0];
  const float* Aev = (const float*)d_in[1];   // complex64 (F,N) -> float pairs
  const float* sBw = (const float*)d_in[2];
  const float* sBb = (const float*)d_in[3];
  const float* sCw = (const float*)d_in[4];
  const float* sCb = (const float*)d_in[5];
  const float* sDw = (const float*)d_in[6];
  const float* sDb = (const float*)d_in[7];
  float* out = (float*)d_out;

  float* ws    = (float*)d_ws;
  float* wsBm  = ws;                              // kML*kN floats
  float* wsD1  = wsBm + (size_t)kML * kN;         // kML floats
  float* wsAB  = wsD1 + kML;                      // kML*kN*4 floats
  float* wsG   = wsAB + (size_t)kML * kN * 4;     // kML*2 floats
  _Float16* wsWh = (_Float16*)(wsG + (size_t)kML * 2);  // 4*32*48 halves (32B aligned)

  s6_wconv<<<1, 128, 0, stream>>>(sBw, sDw, wsWh);
  s6_proj<<<kML / 16, 32, 0, stream>>>(x, wsWh, wsBm, wsD1);
  s6_adbd<<<(kML * kN + 255) / 256, 256, 0, stream>>>(wsBm, wsD1, Aev, sBb, sDb, wsAB);
  s6_scan<<<1, 128, 0, stream>>>(x, sCw, sCb, wsAB, wsG);
  s6_out<<<kB, 128, 0, stream>>>(x, wsG, out);
}
